// CrossAttentionLayer_39788577030788
// MI455X (gfx1250) — matrix operand order repaired and checked by reading, not compiled, not run on hardware
//
#include <hip/hip_runtime.h>
#include <stdint.h>

// ---------------------------------------------------------------------------
// Cross-attention for MI455X (gfx1250, wave32, WMMA).
//   Phase 1: 1x1-conv projections -> f16 Q (natural k order), Kt, V in d_ws.
//   Phase 2: flash-attention, f16 WMMA w/ fp32 accumulate:
//     - K/V tiles double-buffered into LDS via global_load_async_to_lds_*
//       (ASYNCcnt): one WG loads each tile once for all 8 waves.
//     - streaming softmax with per-half-tile running max (exact: softmax is
//       invariant to constant shifts; the reference's global-max subtraction
//       is one too). Row sums accumulated on the MATRIX pipe via an all-ones
//       B operand (accL = P @ 1), not VALU butterflies.
//     - the single cross-lane max uses DPP16 row_ror rotations (pure VALU).
// ---------------------------------------------------------------------------

typedef _Float16 v8h  __attribute__((ext_vector_type(8)));
typedef _Float16 v16h __attribute__((ext_vector_type(16)));
typedef float    v8f  __attribute__((ext_vector_type(8)));

#define B_   4
#define C_   256
#define N_   4096
#define CQK  32

// ---------------- DPP16 row reduction (rows = 16-lane halves) --------------
template <int CTRL>
__device__ __forceinline__ float dpp_rot(float v) {
    // old=0 + bound_ctrl=true -> single v_mov_b32_dpp (no copy of old)
    return __int_as_float(
        __builtin_amdgcn_update_dpp(0, __float_as_int(v), CTRL, 0xF, 0xF, true));
}
__device__ __forceinline__ float rowmax16(float x) {
    x = fmaxf(x, dpp_rot<0x128>(x));   // row_ror:8
    x = fmaxf(x, dpp_rot<0x124>(x));   // row_ror:4
    x = fmaxf(x, dpp_rot<0x122>(x));   // row_ror:2
    x = fmaxf(x, dpp_rot<0x121>(x));   // row_ror:1
    return x;
}

// ---------------- gfx1250 async global->LDS copies (ASYNCcnt) --------------
__device__ __forceinline__ uint32_t lds_off(const void* p) {
    return (uint32_t)(uintptr_t)p;   // low 32 bits of generic ptr = LDS offset
}
__device__ __forceinline__ void async_ld_b128(uint32_t lds, const void* g) {
    asm volatile("global_load_async_to_lds_b128 %0, %1, off"
                 :: "v"(lds), "v"((uint64_t)(uintptr_t)g) : "memory");
}
__device__ __forceinline__ void async_ld_b64(uint32_t lds, const void* g) {
    asm volatile("global_load_async_to_lds_b64 %0, %1, off"
                 :: "v"(lds), "v"((uint64_t)(uintptr_t)g) : "memory");
}
__device__ __forceinline__ void wait_async_le5() {
    asm volatile("s_wait_asynccnt 0x5" ::: "memory");   // prev 5-op batch done
}

// A-operand (16-bit, 16x32) per-lane K order: lanes 0-15 hold K{0..7,16..23},
// lanes 16-31 hold K{8..15,24..31}; pre-permute Q rows so each lane's v16h
// operand is two contiguous B128 loads.
__device__ __forceinline__ int kposA(int o) {
    return ((o >> 3) == 1) ? (o + 8) : (((o >> 3) == 2) ? (o - 8) : o);
}

// load 16 contiguous f16 (32B) as v16h via two B128 loads (global or LDS)
__device__ __forceinline__ v16h ld16(const _Float16* p) {
    v8h lo = *(const v8h*)p;
    v8h hi = *(const v8h*)(p + 8);
    return __builtin_shufflevector(lo, hi, 0,1,2,3,4,5,6,7,8,9,10,11,12,13,14,15);
}

// ---------------------------- Phase 1: Q/K proj ----------------------------
__global__ __launch_bounds__(256) void qk_proj_kernel(
    const float* __restrict__ inp, const float* __restrict__ str,
    const float* __restrict__ Wq, const float* __restrict__ bq,
    const float* __restrict__ Wk, const float* __restrict__ bk,
    _Float16* __restrict__ Qp, _Float16* __restrict__ Kt) {
    const int gid = blockIdx.x * 256 + threadIdx.x;   // one thread per (b, n)
    const int b = gid / N_, n = gid % N_;
    float qa[CQK], ka[CQK];
#pragma unroll
    for (int o = 0; o < CQK; ++o) { qa[o] = bq[o]; ka[o] = bk[o]; }
    const float* xi = inp + (size_t)b * C_ * N_ + n;
    const float* xs = str + (size_t)b * C_ * N_ + n;
    for (int c = 0; c < C_; ++c) {                    // weights uniform -> SMEM
        const float x = xi[(size_t)c * N_];
        const float s = xs[(size_t)c * N_];
#pragma unroll
        for (int o = 0; o < CQK; ++o) {
            qa[o] = fmaf(x, Wq[o * C_ + c], qa[o]);
            ka[o] = fmaf(s, Wk[o * C_ + c], ka[o]);
        }
    }
    _Float16* qrow = Qp + (size_t)(b * N_ + n) * CQK;   // A-layout (natural)
    _Float16* krow = Kt + (size_t)(b * N_ + n) * CQK;   // B-layout (natural)
#pragma unroll
    for (int o = 0; o < CQK; ++o) {
        qrow[o] = (_Float16)qa[o];
        krow[o]        = (_Float16)ka[o];
    }
}

// ---------------------------- Phase 1: V proj ------------------------------
__global__ __launch_bounds__(256) void v_proj_kernel(
    const float* __restrict__ sty, const float* __restrict__ Wv,
    const float* __restrict__ bv, _Float16* __restrict__ Vh) {
    const int c0  = blockIdx.y * 32;                  // 8 output-channel groups
    const int gid = blockIdx.x * 256 + threadIdx.x;   // one thread per (b, n)
    const int b = gid / N_, n = gid % N_;
    float va[32];
#pragma unroll
    for (int o = 0; o < 32; ++o) va[o] = bv[c0 + o];
    const float* xs = sty + (size_t)b * C_ * N_ + n;
    for (int c = 0; c < C_; ++c) {
        const float s = xs[(size_t)c * N_];
#pragma unroll
        for (int o = 0; o < 32; ++o)
            va[o] = fmaf(s, Wv[(c0 + o) * C_ + c], va[o]);
    }
#pragma unroll
    for (int o = 0; o < 32; ++o)
        Vh[((size_t)b * C_ + c0 + o) * N_ + n] = (_Float16)va[o];
}

// ------------------------- Phase 2: flash attention ------------------------
// Grid: B * (N/64) blocks of 256 threads (8 wave32).
// Wave w: query rows [qtile*64 + (w&3)*16, +16), V columns [(w>>2)*128, +128).
__global__ __launch_bounds__(256) void flash_attn_kernel(
    const _Float16* __restrict__ Qp, const _Float16* __restrict__ Kt,
    const _Float16* __restrict__ Vh, float* __restrict__ out) {
    __shared__ alignas(16) _Float16 sK[2][32 * CQK];   //  2x 2KB  key block
    __shared__ alignas(16) _Float16 sV[2][C_ * 32];    //  2x 16KB value block
    __shared__ alignas(16) _Float16 sP[8][16 * 32];    //  per-wave P tiles

    const int tid  = threadIdx.x;
    const int w    = tid >> 5;
    const int lane = tid & 31;
    const int li   = lane & 15;
    const int h    = lane >> 4;                        // WMMA operand lane-half
    const int b    = blockIdx.x >> 6;
    const int qt   = blockIdx.x & 63;
    const int qbase = qt * 64 + (w & 3) * 16;
    const int c0    = (w >> 2) * 128;

    const _Float16* Kb = Kt + (size_t)b * N_ * CQK;
    const _Float16* Vb = Vh + (size_t)b * C_ * N_;
    _Float16* pbuf = &sP[w][0];

    // Uniform 5-op async batch per thread: 1x b64 (K 2KB) + 4x b128 (V 16KB).
    auto issue_batch = [&](int buf, int j) {
        async_ld_b64(lds_off(&sK[buf][0]) + tid * 8,
                     Kb + (size_t)(j + (tid >> 3)) * CQK + (tid & 7) * 4);
        const _Float16* g = Vb + (size_t)tid * N_ + j;   // c-row = tid
        const uint32_t  l = lds_off(&sV[buf][tid * 32]);
#pragma unroll
        for (int k = 0; k < 4; ++k)
            async_ld_b128(l + k * 16, g + k * 8);
    };

    // Loop-invariant Q A-operand: row m = li, K-half h (natural storage, k = 16h + s)
    const v16h a_q = ld16(Qp + (size_t)(b * N_ + qbase + li) * CQK + h * 16);

    // All-ones B operand: accL = P @ 1 accumulates per-row sum(exp) on the
    // matrix pipe (every lane's element == its row's sum).
    v16h ones;
#pragma unroll
    for (int i = 0; i < 16; ++i) ones[i] = (_Float16)1.0f;

    v8f acc[8];
    const v8f vzero = {};
#pragma unroll
    for (int t = 0; t < 8; ++t) acc[t] = vzero;
    v8f accL = vzero;                  // running sum(exp) per row
    float rm = -1e30f;                 // running max of this half's 16x... tile

    issue_batch(0, 0);                                 // prime buffer 0

    for (int i = 0; i < N_ / 32; ++i) {
        const int cur = i & 1;
        int jn = 32 * i + 32;
        if (jn >= N_) jn = 32 * i;                     // tail: harmless reload
        issue_batch(cur ^ 1, jn);                      // prefetch next block
        wait_async_le5();                              // my previous batch done
        __syncthreads();                               // whole tile published

        // S = Q(16x32) @ K(32c x 32keys) from LDS. B-operand: lane = key col,
        // contiguous 16-element K-half per lane group.
        const v16h b_k0 = ld16(&sK[cur][(size_t)li * CQK + h * 16]);
        const v16h b_k1 = ld16(&sK[cur][(size_t)(li + 16) * CQK + h * 16]);
        v8f s0 = __builtin_amdgcn_wmma_f32_16x16x32_f16(false, a_q, false, b_k0,
                                                        (short)0, vzero, false, false);
        v8f s1 = __builtin_amdgcn_wmma_f32_16x16x32_f16(false, a_q, false, b_k1,
                                                        (short)0, vzero, false, false);

        // Half-tile running max (bound so exp(P) <= 1 fits f16); a constant
        // shift per half-tile keeps softmax exact.
        float tm = fmaxf(s0[0], s1[0]);
#pragma unroll
        for (int r = 1; r < 8; ++r) tm = fmaxf(tm, fmaxf(s0[r], s1[r]));
        tm = rowmax16(tm);                       // uniform across the 16 lanes
        const float mnew = fmaxf(rm, tm);
        const float sc = __expf(rm - mnew);      // lane-uniform rescale factor
        rm = mnew;

#pragma unroll
        for (int t = 0; t < 8; ++t)
#pragma unroll
            for (int r = 0; r < 8; ++r) acc[t][r] *= sc;
#pragma unroll
        for (int r = 0; r < 8; ++r) accL[r] *= sc;

        float p0[8], p1[8];
#pragma unroll
        for (int r = 0; r < 8; ++r) {
            p0[r] = __expf(s0[r] - mnew);
            p1[r] = __expf(s1[r] - mnew);
        }

        // P: C-layout -> per-wave LDS tile (row-major [m][kk]) -> A-operand.
        // Same-wave DS ops are in-order; no barrier needed for this buffer.
#pragma unroll
        for (int r = 0; r < 8; ++r) {
            pbuf[(r + 8 * h) * 32 + li]      = (_Float16)p0[r];
            pbuf[(r + 8 * h) * 32 + li + 16] = (_Float16)p1[r];
        }
        v8h plo = *(const v8h*)(pbuf + li * 32 + h * 16);       // K {0..7}/{16..23}
        v8h phi = *(const v8h*)(pbuf + li * 32 + h * 16 + 8);   // K {8..15}/{24..31}
        const v16h a_p = __builtin_shufflevector(plo, phi,
                           0,1,2,3,4,5,6,7,8,9,10,11,12,13,14,15);

        // Row sums on the matrix pipe + acc += P @ V (8 column tiles).
        accL = __builtin_amdgcn_wmma_f32_16x16x32_f16(false, a_p, false, ones,
                                                      (short)0, accL, false, false);
#pragma unroll
        for (int t = 0; t < 8; ++t) {
            const v16h b_v = ld16(&sV[cur][(size_t)(c0 + t * 16 + li) * 32 + h * 16]);
            acc[t] = __builtin_amdgcn_wmma_f32_16x16x32_f16(false, a_p, false, b_v,
                                                            (short)0, acc[t], false, false);
        }
        __syncthreads();   // all waves done with buf[cur] before it is refilled
    }

    // Epilogue: out[b, c, n] = acc / sum(exp).
#pragma unroll
    for (int r = 0; r < 8; ++r) {
        const float inv = 1.0f / accL[r];
        const int n_out = qbase + r + 8 * h;
#pragma unroll
        for (int t = 0; t < 8; ++t) {
            const int c = c0 + t * 16 + li;
            out[((size_t)b * C_ + c) * N_ + n_out] = acc[t][r] * inv;
        }
    }
}

// ---------------------------------------------------------------------------
extern "C" void kernel_launch(void* const* d_in, const int* in_sizes, int n_in,
                              void* d_out, int out_size, void* d_ws, size_t ws_size,
                              hipStream_t stream) {
    (void)in_sizes; (void)n_in; (void)out_size; (void)ws_size;
    const float* input     = (const float*)d_in[0];
    const float* structure = (const float*)d_in[1];
    const float* style     = (const float*)d_in[2];
    const float* Wq        = (const float*)d_in[3];
    const float* bq        = (const float*)d_in[4];
    const float* Wk        = (const float*)d_in[5];
    const float* bk        = (const float*)d_in[6];
    const float* Wv        = (const float*)d_in[7];
    const float* bv        = (const float*)d_in[8];
    float* out             = (float*)d_out;

    // Workspace carve (f16): Qp 1MB | Kt 1MB | Vh 8MB   (all 16B-aligned)
    _Float16* Qp = (_Float16*)d_ws;
    _Float16* Kt = Qp + (size_t)B_ * N_ * CQK;
    _Float16* Vh = Kt + (size_t)B_ * N_ * CQK;

    qk_proj_kernel<<<dim3(B_ * N_ / 256), dim3(256), 0, stream>>>(
        input, structure, Wq, bq, Wk, bk, Qp, Kt);
    v_proj_kernel<<<dim3(B_ * N_ / 256, 8), dim3(256), 0, stream>>>(
        style, Wv, bv, Vh);
    flash_attn_kernel<<<dim3(B_ * (N_ / 64)), dim3(256), 0, stream>>>(
        Qp, Kt, Vh, out);
}
